// CorrBlock_19370302505632
// MI455X (gfx1250) — matrix operand order repaired and checked by reading, not compiled, not run on hardware
//
#include <hip/hip_runtime.h>

// Problem constants (fixed by reference setup_inputs)
#define BB   4
#define CC   256
#define HH   96
#define WW   320
#define HWc  (HH * WW)          // 30720, channel stride (elements)
#define BSTR (CC * HWc)         // batch stride in fmaps
#define NROW (BB * HH)          // 384 row-GEMMs
#define KTILE 32

typedef __attribute__((ext_vector_type(16))) __bf16 v16bf_t;
typedef __attribute__((ext_vector_type(8)))  float  v8f_t;
typedef __attribute__((ext_vector_type(4)))  unsigned int u32x4_t;
typedef __attribute__((ext_vector_type(8)))  int    i32x8_t;
typedef __attribute__((ext_vector_type(4)))  int    i32x4_t;

// ---------------------------------------------------------------------------
// TDM: async-load one 32(K) x 64(x) f32 tile (row stride HWc elements) from
// global memory into LDS at byte offset lds_off, laid out [k][64x] contiguous.
// D# descriptor per CDNA5 ISA sec. 8 (2D tensor, groups 2/3 unused).
// Toolchain here is the 6-arg builtin variant (clang-23 / therock-10.0).
// ---------------------------------------------------------------------------
__device__ __forceinline__ void tdm_load_tile_f32(unsigned int lds_off,
                                                  const float* gsrc) {
    const unsigned long long ga = (unsigned long long)(uintptr_t)gsrc;
    u32x4_t g0;
    g0[0] = 1u;                                   // count=1, user descriptor
    g0[1] = lds_off;                              // lds_addr (bytes)
    g0[2] = (unsigned int)(ga & 0xffffffffull);   // global_addr[31:0]
    g0[3] = (unsigned int)((ga >> 32) & 0x01ffffffull) | (2u << 30); // [56:32], type=2

    i32x8_t g1;
    g1[0] = (int)(2u << 16);                      // data_size=4B; no pad/iter/mask
    g1[1] = (int)(((unsigned)WW & 0xffffu) << 16);      // tensor_dim0=320 (lo16)
    g1[2] = (int)((((unsigned)WW >> 16) & 0xffffu) |    // tensor_dim0 hi
                  (((unsigned)CC & 0xffffu) << 16));    // tensor_dim1=256 (lo16)
    g1[3] = (int)((((unsigned)CC >> 16) & 0xffffu) |    // tensor_dim1 hi
                  (64u << 16));                         // tile_dim0=64
    g1[4] = (int)(32u);                           // tile_dim1=32, tile_dim2=0
    g1[5] = (int)((unsigned)HWc);                 // tensor_dim0_stride[31:0]=30720
    g1[6] = 0;                                    // stride hi, dim1_stride lo
    g1[7] = 0;

    const i32x4_t z4 = {0, 0, 0, 0};              // groups 2/3 unused (2D)
    const i32x8_t z8 = {0, 0, 0, 0, 0, 0, 0, 0};  // extended group unused
    __builtin_amdgcn_tensor_load_to_lds(g0, g1, z4, z4, z8, 0);
}

// ---------------------------------------------------------------------------
// Kernel 1: per-(b,row) GEMM  corr[n][m][v] = sum_c f1[n][c][m]*f2[n][c][v]/16
// Block = 128 threads (4 waves); 64x64 f32 tile; each wave owns a 32x32
// quadrant (2x2 WMMA tiles). K pipelined in 32-chunks via TDM double buffer.
// grid = (WW/64, WW/64, NROW) = (5, 5, 384)
// ---------------------------------------------------------------------------
__global__ __launch_bounds__(128)
void corr_wmma_kernel(const float* __restrict__ fmap1,
                      const float* __restrict__ fmap2,
                      float* __restrict__ corr) {
    // [pingpong][A=0/B=1][k*64 + x]  f32, 4 x 8KB = 32KB
    __shared__ float sBuf[2][2][KTILE * 64];

    const int n  = blockIdx.z;              // 0..383 = b*HH + y
    const int b  = n / HH;
    const int y  = n - b * HH;
    const int M0 = blockIdx.y * 64;
    const int N0 = blockIdx.x * 64;

    const size_t base = (size_t)b * BSTR + (size_t)y * WW;
    const float* __restrict__ aPtr = fmap1 + base + M0;  // (c,x): c*HWc + x
    const float* __restrict__ bPtr = fmap2 + base + N0;

    const int tid  = threadIdx.x;
    const int lane = tid & 31;
    const int wid  = tid >> 5;              // 0..3
    const int mt   = wid >> 1;              // 0/1 : 32-row half of M tile
    const int nt   = wid & 1;               // 0/1 : 32-col half of N tile

    // Fragment addressing per CDNA5 ISA 7.12.2 (wave32), bf16 16x16x32:
    // A: lanes 0-15 hold K{0..7,16..23}; lanes 16-31 hold K{8..15,24..31}
    // B: lanes 0-15 hold K{0..7,16..23}; lanes 16-31 hold K{8..15,24..31}
    const int kbA = (lane >> 4) * 8;
    const int kbB = (lane >> 4) * 16;
    const int rA0 = mt * 32 + (lane & 15);  // A spatial row of first subtile
    const int cB0 = nt * 32 + (lane & 15);  // B spatial col of first subtile

    v8f_t acc00 = {}, acc01 = {}, acc10 = {}, acc11 = {};

    // Prologue: TDM chunk 0 into buffer 0 (one wave issues; EXEC-independent).
    if (tid < 32) {
        tdm_load_tile_f32((unsigned int)(uintptr_t)&sBuf[0][0][0], aPtr);
        tdm_load_tile_f32((unsigned int)(uintptr_t)&sBuf[0][1][0], bPtr);
    }

    #pragma unroll 2
    for (int kc = 0; kc < CC / KTILE; ++kc) {
        if (tid < 32) {
            if (kc < CC / KTILE - 1) {
                const size_t koff = (size_t)(kc + 1) * KTILE * HWc;
                const int nb = (kc + 1) & 1;
                tdm_load_tile_f32((unsigned int)(uintptr_t)&sBuf[nb][0][0], aPtr + koff);
                tdm_load_tile_f32((unsigned int)(uintptr_t)&sBuf[nb][1][0], bPtr + koff);
                // in-order TENSORcnt: <=2 outstanding => current chunk landed
                __builtin_amdgcn_s_wait_tensorcnt(2);
            } else {
                __builtin_amdgcn_s_wait_tensorcnt(0);
            }
        }
        __syncthreads();

        const int p = kc & 1;
        v16bf_t a0, a1, b0, b1;
        #pragma unroll
        for (int i = 0; i < 8; ++i) {
            a0[i]     = (__bf16)sBuf[p][0][(kbA + i) * 64 + rA0];
            a0[i + 8] = (__bf16)sBuf[p][0][(kbA + 16 + i) * 64 + rA0];
            a1[i]     = (__bf16)sBuf[p][0][(kbA + i) * 64 + rA0 + 16];
            a1[i + 8] = (__bf16)sBuf[p][0][(kbA + 16 + i) * 64 + rA0 + 16];
        }
        #pragma unroll
        for (int i = 0; i < 8; ++i) {
            b0[i]     = (__bf16)sBuf[p][1][(kbA + i) * 64 + cB0];
            b0[i + 8] = (__bf16)sBuf[p][1][(kbA + 16 + i) * 64 + cB0];
            b1[i]     = (__bf16)sBuf[p][1][(kbA + i) * 64 + cB0 + 16];
            b1[i + 8] = (__bf16)sBuf[p][1][(kbA + 16 + i) * 64 + cB0 + 16];
        }

        acc00 = __builtin_amdgcn_wmma_f32_16x16x32_bf16(false, a0, false, b0, (short)0, acc00, false, false);
        acc01 = __builtin_amdgcn_wmma_f32_16x16x32_bf16(false, a0, false, b1, (short)0, acc01, false, false);
        acc10 = __builtin_amdgcn_wmma_f32_16x16x32_bf16(false, a1, false, b0, (short)0, acc10, false, false);
        acc11 = __builtin_amdgcn_wmma_f32_16x16x32_bf16(false, a1, false, b1, (short)0, acc11, false, false);
        __syncthreads();
    }

    // C/D layout: lanes 0-15: vgpr r -> (M=r, N=lane); lanes 16-31: (M=8+r, N=lane-16)
    const int mB = M0 + mt * 32 + (lane >> 4) * 8;
    const int nB = N0 + nt * 32 + (lane & 15);
    float* __restrict__ crow = corr + (size_t)n * WW * WW;
    #pragma unroll
    for (int r = 0; r < 8; ++r) {
        crow[(size_t)(mB + r) * WW + nB]           = acc00[r] * 0.0625f;  // /sqrt(256)
        crow[(size_t)(mB + r) * WW + nB + 16]      = acc01[r] * 0.0625f;
        crow[(size_t)(mB + 16 + r) * WW + nB]      = acc10[r] * 0.0625f;
        crow[(size_t)(mB + 16 + r) * WW + nB + 16] = acc11[r] * 0.0625f;
    }
}

// ---------------------------------------------------------------------------
// Kernel 2: pyramid pooling + 9-tap bilinear lookup, fused.
// One thread per query pixel; its 320-float corr row is L2-resident (157 MB
// volume < 192 MB L2). Level-l sample value = mean of 2^l level-0 elements.
// ---------------------------------------------------------------------------
__device__ __forceinline__ float pool_val(const float* __restrict__ row,
                                          int i, int Wl, int p, float inv_p) {
    if (i < 0 || i >= Wl) return 0.0f;   // zeros padding, matches reference
    float s = 0.0f;
    #pragma unroll 8
    for (int d = 0; d < p; ++d) s += row[i * p + d];
    return s * inv_p;
}

__global__ __launch_bounds__(256)
void corr_lookup_kernel(const float* __restrict__ corr,
                        const float* __restrict__ centroids,
                        float* __restrict__ out) {
    const int pix = blockIdx.x * blockDim.x + threadIdx.x;  // (b*HH+y)*WW + x
    if (pix >= BB * HH * WW) return;
    const int x   = pix % WW;
    const int rem = pix / WW;
    const int y   = rem % HH;
    const int b   = rem / HH;

    const float* __restrict__ row = corr + (size_t)pix * WW;
    const float cent = centroids[pix];

    #pragma unroll
    for (int lvl = 0; lvl < 4; ++lvl) {
        const int p      = 1 << lvl;
        const int Wl     = WW >> lvl;
        const float invp = 1.0f / (float)p;
        const float cl   = cent * invp;
        #pragma unroll
        for (int j = 0; j < 9; ++j) {
            const float xx = cl + (float)(j - 4);
            const float x0 = floorf(xx);
            const float t  = xx - x0;
            const int   i0 = (int)x0;
            const float v0 = pool_val(row, i0,     Wl, p, invp);
            const float v1 = pool_val(row, i0 + 1, Wl, p, invp);
            const int ch = lvl * 9 + j;
            out[((size_t)(b * 36 + ch) * HH + y) * WW + x] = v0 * (1.0f - t) + v1 * t;
        }
    }
}

// ---------------------------------------------------------------------------
extern "C" void kernel_launch(void* const* d_in, const int* in_sizes, int n_in,
                              void* d_out, int out_size, void* d_ws, size_t ws_size,
                              hipStream_t stream) {
    const float* fmap1 = (const float*)d_in[0];
    const float* fmap2 = (const float*)d_in[1];
    const float* cent  = (const float*)d_in[2];
    float* out  = (float*)d_out;
    float* corr = (float*)d_ws;   // needs 384*320*320*4 = 157,286,400 bytes

    dim3 gA(WW / 64, WW / 64, NROW);   // (5,5,384)
    corr_wmma_kernel<<<gA, 128, 0, stream>>>(fmap1, fmap2, corr);

    const int total = BB * HH * WW;    // 122880
    corr_lookup_kernel<<<(total + 255) / 256, 256, 0, stream>>>(corr, cent, out);
}
